// PretrainedALIGNN_85676007621243
// MI455X (gfx1250) — compile-verified, matrix operand order checked by reading, not run on hardware
//
#include <hip/hip_runtime.h>
#include <hip/hip_bf16.h>
#include <stdint.h>

// ---------------------------------------------------------------------------
// ALIGNN forward for MI455X (gfx1250, wave32, WMMA).
//
// Input order assumption: harness flattens the setup_inputs() dict as a JAX
// pytree (dict keys sorted alphabetically at every level):
//   0 atom_feats[4096,92]  1 dst[49152]  2 graph_ids[4096]
//   3 lg_dst[262144]       4 lg_src[262144]
//   5..8   angle1 {W[40,64], b, beta, g}
//   9..12  angle2 {W[64,256], b, beta, g}
//   13..16 atom   {W[92,256], b, beta, g}
//   17 conv_W[12,5,256,256] 18 conv_b[12,5,256]
//   19 conv_lng[12,2,256]   20 conv_lnb[12,2,256]
//   21..24 edge1 {W[80,64], b, beta, g}
//   25..28 edge2 {W[64,256], b, beta, g}
//   29 fc.W1[256,256] 30 fc.W2[256,128] 31 fc.W3[128,1]
//   32 fc.b1 33 fc.b2 34 fc.b3 35 fc.beta1 36 fc.beta2 37 fc.g1 38 fc.g2
//   39 r[49152,3] 40 src[49152]
// ---------------------------------------------------------------------------

#define N_NODES 4096
#define N_EDGES 49152
#define N_TRIP  262144
#define N_GRAPH 64
#define HID 256

typedef __attribute__((ext_vector_type(16))) __bf16 v16bf;
typedef __attribute__((ext_vector_type(8)))  float  v8f;

__device__ __forceinline__ unsigned short f2bf(float f) {
  union { float f; unsigned u; } v; v.f = f;
  unsigned u = v.u + 0x7FFFu + ((v.u >> 16) & 1u);   // round-to-nearest-even
  return (unsigned short)(u >> 16);
}

// ---------------------------------------------------------------------------
// WMMA GEMM:  Out[rows,N] = X[rows,K] * W[K,N] + bias[N]   (fp32 in/out,
// bf16 multiply, f32 accumulate). Block tile 128x128, BK=32, 256 threads
// (8 waves = 4M x 2N), each wave computes 32x64 via 2x4 WMMA fragments.
// LDS tiles are stored pre-swizzled in WMMA fragment layout:
//   fragment-major, each lane's 16 bf16 elements contiguous (32 bytes).
// A 16x32 bf16 fragment layout (ISA 7.12.2):
//   lane = (row%16) + 16*((K>>3)&1),  elem = (K&7) + 8*((K>>4)&1)
// B 32x16 uses the symmetric mapping with col in place of row.
// ---------------------------------------------------------------------------

// Guarded (boundary-tile) stager: per-element range checks.
__device__ __forceinline__ void stage_tile_guard(
    unsigned short* sA, unsigned short* sB,
    const float* __restrict__ X, const float* __restrict__ W,
    int rows, int K, int N, int mBase, int nBase, int kBase, int t)
{
  {
    int rlocal = t >> 1;
    int kh     = t & 1;
    int frag   = rlocal >> 4;
    int rr     = rlocal & 15;
    int grow   = mBase + rlocal;
    const float* Xr = X + (size_t)grow * K;
#pragma unroll
    for (int kk = 0; kk < 16; kk += 2) {
      int Kl = kh * 16 + kk;
      int gk = kBase + Kl;
      float f0 = (grow < rows && gk     < K) ? Xr[gk]     : 0.f;
      float f1 = (grow < rows && gk + 1 < K) ? Xr[gk + 1] : 0.f;
      int khi  = (Kl >> 3) & 1;
      int e    = (Kl & 7) + 8 * ((Kl >> 4) & 1);
      int lt   = rr + 16 * khi;
      unsigned pack = (unsigned)f2bf(f0) | ((unsigned)f2bf(f1) << 16);
      *(unsigned*)&sA[frag * 512 + lt * 16 + e] = pack;
    }
  }
  {
    int Kl   = t & 31;
    int frag = t >> 5;
    int gk   = kBase + Kl;
    int khi  = (Kl >> 3) & 1;
    int e    = (Kl & 7) + 8 * ((Kl >> 4) & 1);
    const float* Wr = W + (size_t)gk * N;
#pragma unroll
    for (int c = 0; c < 16; ++c) {
      int gcol = nBase + frag * 16 + c;
      float f = (gk < K && gcol < N) ? Wr[gcol] : 0.f;
      sB[frag * 512 + (c + 16 * khi) * 16 + e] = f2bf(f);
    }
  }
}

// Fast (interior-tile) stager: unguarded vectorized float4 loads
// (global_load_b128) + prefetch of the next K tile.
__device__ __forceinline__ void stage_tile_fast(
    unsigned short* sA, unsigned short* sB,
    const float* __restrict__ X, const float* __restrict__ W,
    int K, int N, int mBase, int nBase, int kBase, int t)
{
  {
    int rlocal = t >> 1, kh = t & 1;
    int frag = rlocal >> 4, rr = rlocal & 15;
    const float* Xp = X + (size_t)(mBase + rlocal) * K + kBase + kh * 16;
    float a[16];
    *(float4*)&a[0]  = *(const float4*)&Xp[0];
    *(float4*)&a[4]  = *(const float4*)&Xp[4];
    *(float4*)&a[8]  = *(const float4*)&Xp[8];
    *(float4*)&a[12] = *(const float4*)&Xp[12];
    __builtin_prefetch(Xp + 32, 0, 1);           // next K tile
#pragma unroll
    for (int kk = 0; kk < 16; kk += 2) {
      int Kl  = kh * 16 + kk;
      int khi = (Kl >> 3) & 1;
      int e   = (Kl & 7) + 8 * ((Kl >> 4) & 1);
      int lt  = rr + 16 * khi;
      unsigned pack = (unsigned)f2bf(a[kk]) | ((unsigned)f2bf(a[kk + 1]) << 16);
      *(unsigned*)&sA[frag * 512 + lt * 16 + e] = pack;
    }
  }
  {
    int Kl = t & 31, frag = t >> 5;
    int khi = (Kl >> 3) & 1;
    int e   = (Kl & 7) + 8 * ((Kl >> 4) & 1);
    const float* Wp = W + (size_t)(kBase + Kl) * N + nBase + frag * 16;
    float b[16];
    *(float4*)&b[0]  = *(const float4*)&Wp[0];
    *(float4*)&b[4]  = *(const float4*)&Wp[4];
    *(float4*)&b[8]  = *(const float4*)&Wp[8];
    *(float4*)&b[12] = *(const float4*)&Wp[12];
    __builtin_prefetch(Wp + 32 * N, 0, 1);       // next K tile
#pragma unroll
    for (int c = 0; c < 16; ++c)
      sB[frag * 512 + (c + 16 * khi) * 16 + e] = f2bf(b[c]);
  }
}

__global__ __launch_bounds__(256) void k_gemm_bias(
    const float* __restrict__ X, const float* __restrict__ W,
    const float* __restrict__ bias, float* __restrict__ Out,
    int rows, int K, int N)
{
  __shared__ __attribute__((aligned(32))) unsigned short lA[2][8 * 512];
  __shared__ __attribute__((aligned(32))) unsigned short lB[2][8 * 512];

  const int t     = threadIdx.x;
  const int mBase = blockIdx.y * 128;
  const int nBase = blockIdx.x * 128;
  const int kTiles = (K + 31) / 32;

  const int lane = t & 31;
  const int wid  = t >> 5;
  const int mW   = wid & 3;   // 0..3  -> rows mW*32 .. +32
  const int nW   = wid >> 2;  // 0..1  -> cols nW*64 .. +64

  const bool interior = (mBase + 128 <= rows) && (nBase + 128 <= N);

  v8f acc[2][4];
#pragma unroll
  for (int i = 0; i < 2; ++i)
#pragma unroll
    for (int j = 0; j < 4; ++j) acc[i][j] = {};

  auto stage = [&](int b, int kb) {
    if (interior && kb + 32 <= K)
      stage_tile_fast(lA[b], lB[b], X, W, K, N, mBase, nBase, kb, t);
    else
      stage_tile_guard(lA[b], lB[b], X, W, rows, K, N, mBase, nBase, kb, t);
  };

  stage(0, 0);
  __syncthreads();

  int buf = 0;
  for (int kt = 0; kt < kTiles; ++kt) {
    if (kt + 1 < kTiles) stage(buf ^ 1, (kt + 1) * 32);

    v16bf a0 = *(const v16bf*)&lA[buf][(mW * 2 + 0) * 512 + lane * 16];
    v16bf a1 = *(const v16bf*)&lA[buf][(mW * 2 + 1) * 512 + lane * 16];
#pragma unroll
    for (int j = 0; j < 4; ++j) {
      v16bf b = *(const v16bf*)&lB[buf][(nW * 4 + j) * 512 + lane * 16];
      acc[0][j] = __builtin_amdgcn_wmma_f32_16x16x32_bf16(
          false, a0, false, b, (short)0, acc[0][j], false, false);
      acc[1][j] = __builtin_amdgcn_wmma_f32_16x16x32_bf16(
          false, a1, false, b, (short)0, acc[1][j], false, false);
    }
    __syncthreads();
    buf ^= 1;
  }

  // Epilogue. C/D layout: lanes 0-15 -> M=vg, lanes 16-31 -> M=vg+8; N=lane%16.
  if (interior) {
#pragma unroll
    for (int i = 0; i < 2; ++i) {
      int rowf = mBase + (mW * 2 + i) * 16 + 8 * (lane >> 4);
#pragma unroll
      for (int j = 0; j < 4; ++j) {
        int col = nBase + (nW * 4 + j) * 16 + (lane & 15);
        float bv = bias[col];
#pragma unroll
        for (int vg = 0; vg < 8; ++vg)
          Out[(size_t)(rowf + vg) * N + col] = acc[i][j][vg] + bv;
      }
    }
  } else {
#pragma unroll
    for (int i = 0; i < 2; ++i) {
      int rowf = mBase + (mW * 2 + i) * 16 + 8 * (lane >> 4);
#pragma unroll
      for (int j = 0; j < 4; ++j) {
        int col = nBase + (nW * 4 + j) * 16 + (lane & 15);
        float bv = (col < N) ? bias[col] : 0.f;
#pragma unroll
        for (int vg = 0; vg < 8; ++vg) {
          int row = rowf + vg;
          if (row < rows && col < N)
            Out[(size_t)row * N + col] = acc[i][j][vg] + bv;
        }
      }
    }
  }
}

// ---------------------------------------------------------------------------
// Elementwise / graph kernels (fp32, bandwidth-bound)
// ---------------------------------------------------------------------------
__global__ void k_zero(float* p, long long n) {
  long long i = (long long)blockIdx.x * blockDim.x + threadIdx.x;
  long long s = (long long)gridDim.x * blockDim.x;
  for (; i < n; i += s) p[i] = 0.f;
}

__global__ void k_bond_rbf(const float* __restrict__ r, float* __restrict__ out, int E) {
  int e = blockIdx.x * blockDim.x + threadIdx.x;
  if (e >= E) return;
  float x = r[e * 3], y = r[e * 3 + 1], z = r[e * 3 + 2];
  float len = sqrtf(x * x + y * y + z * z);
  const float gamma = 79.f / 8.f;
#pragma unroll 8
  for (int b = 0; b < 80; ++b) {
    float c = 8.f * (float)b / 79.f;
    float d = len - c;
    out[(size_t)e * 80 + b] = __expf(-gamma * d * d);
  }
}

__global__ void k_angle_rbf(const float* __restrict__ r,
                            const int* __restrict__ ls, const int* __restrict__ ld,
                            float* __restrict__ out, int T) {
  int t = blockIdx.x * blockDim.x + threadIdx.x;
  if (t >= T) return;
  int a = ls[t], b2 = ld[t];
  float ax = -r[a * 3], ay = -r[a * 3 + 1], az = -r[a * 3 + 2];
  float bx = r[b2 * 3], by = r[b2 * 3 + 1], bz = r[b2 * 3 + 2];
  float dot = ax * bx + ay * by + az * bz;
  float n1 = sqrtf(ax * ax + ay * ay + az * az);
  float n2 = sqrtf(bx * bx + by * by + bz * bz);
  float cs = dot / (n1 * n2 + 1e-12f);
  cs = fminf(1.f, fmaxf(-1.f, cs));
  const float gamma = 39.f / 2.f;
#pragma unroll 8
  for (int b = 0; b < 40; ++b) {
    float c = -1.f + 2.f * (float)b / 39.f;
    float d = cs - c;
    out[(size_t)t * 40 + b] = __expf(-gamma * d * d);
  }
}

// out = [res +] act( LN( in [+ sumSH/(sumS+1e-5)] ) * g + beta )
// One wave per row (wave32). d in {64,128,256}. act: 0=silu, 1=relu.
__global__ __launch_bounds__(256) void k_ln_act(
    const float* __restrict__ in, const float* sumSH, const float* sumS,
    const float* res, const float* __restrict__ g, const float* __restrict__ beta,
    float* __restrict__ out, int rows, int d, int act)
{
  int wid = blockIdx.x * (blockDim.x >> 5) + (threadIdx.x >> 5);
  if (wid >= rows) return;
  int lane = threadIdx.x & 31;
  int npl = d >> 5;                 // elems per lane: 2/4/8
  float v[8];
  float s = 0.f, ss = 0.f;
  size_t base = (size_t)wid * d;
  for (int i = 0; i < npl; ++i) {
    int j = lane + i * 32;
    float x = in[base + j];
    if (sumSH) x += sumSH[base + j] / (sumS[base + j] + 1e-5f);
    v[i] = x; s += x; ss += x * x;
  }
  for (int o = 16; o; o >>= 1) { s += __shfl_xor(s, o, 32); ss += __shfl_xor(ss, o, 32); }
  float mean = s / (float)d;
  float var  = ss / (float)d - mean * mean;
  float rinv = rsqrtf(var + 1e-5f);
  for (int i = 0; i < npl; ++i) {
    int j = lane + i * 32;
    float x = (v[i] - mean) * rinv * g[j] + beta[j];
    if (act == 0)      x = x / (1.f + __expf(-x));   // SiLU
    else if (act == 1) x = fmaxf(x, 0.f);            // ReLU
    if (res) x += res[base + j];
    out[base + j] = x;
  }
}

// m[t,:] = A[src[t],:] + B[dst[t],:] + C[t,:]   (d == 256; out may alias C)
// One wave per edge row, float4 vectorized (global_load_b128).
__global__ __launch_bounds__(256) void k_gather_add3(
    const float4* __restrict__ A, const float4* __restrict__ B,
    const float4* C, const int* __restrict__ src, const int* __restrict__ dst,
    float4* out, int rowsE)
{
  int t = blockIdx.x * 8 + (threadIdx.x >> 5);
  if (t >= rowsE) return;
  int lane = threadIdx.x & 31;
  size_t sb = (size_t)src[t] * 64, db = (size_t)dst[t] * 64, eb = (size_t)t * 64;
#pragma unroll
  for (int i = 0; i < 2; ++i) {
    int j = lane + i * 32;
    float4 a = A[sb + j], b = B[db + j], c = C[eb + j];
    out[eb + j] = make_float4(a.x + b.x + c.x, a.y + b.y + c.y,
                              a.z + b.z + c.z, a.w + b.w + c.w);
  }
}

// sigma = sigmoid(m); sumSH[dst] += sigma * T4[src]; sumS[dst] += sigma
__global__ __launch_bounds__(256) void k_scatter_gates(
    const float4* __restrict__ m, const float4* __restrict__ T4,
    const int* __restrict__ src, const int* __restrict__ dst,
    float* sumSH, float* sumS, int rowsE)
{
  int t = blockIdx.x * 8 + (threadIdx.x >> 5);
  if (t >= rowsE) return;
  int lane = threadIdx.x & 31;
  size_t sb = (size_t)src[t] * 64, eb = (size_t)t * 64;
  size_t db = (size_t)dst[t] * 256;
#pragma unroll
  for (int i = 0; i < 2; ++i) {
    int j = lane + i * 32;
    float4 mv = m[eb + j];
    float4 bh = T4[sb + j];
    float s0 = 1.f / (1.f + __expf(-mv.x));
    float s1 = 1.f / (1.f + __expf(-mv.y));
    float s2 = 1.f / (1.f + __expf(-mv.z));
    float s3 = 1.f / (1.f + __expf(-mv.w));
    size_t col = db + (size_t)j * 4;
    atomicAdd(&sumSH[col + 0], s0 * bh.x); atomicAdd(&sumS[col + 0], s0);
    atomicAdd(&sumSH[col + 1], s1 * bh.y); atomicAdd(&sumS[col + 1], s1);
    atomicAdd(&sumSH[col + 2], s2 * bh.z); atomicAdd(&sumS[col + 2], s2);
    atomicAdd(&sumSH[col + 3], s3 * bh.w); atomicAdd(&sumS[col + 3], s3);
  }
}

__global__ __launch_bounds__(256) void k_pool(
    const float4* __restrict__ x, const int* __restrict__ gid,
    float* hg, float* cnt, int n)
{
  int node = blockIdx.x * 8 + (threadIdx.x >> 5);
  if (node >= n) return;
  int lane = threadIdx.x & 31;
  int g = gid[node];
  size_t nb = (size_t)node * 64, gb = (size_t)g * 256;
#pragma unroll
  for (int i = 0; i < 2; ++i) {
    int j = lane + i * 32;
    float4 v = x[nb + j];
    size_t col = gb + (size_t)j * 4;
    atomicAdd(&hg[col + 0], v.x);
    atomicAdd(&hg[col + 1], v.y);
    atomicAdd(&hg[col + 2], v.z);
    atomicAdd(&hg[col + 3], v.w);
  }
  if (lane == 0) atomicAdd(&cnt[g], 1.f);
}

__global__ void k_pool_div(float* hg, const float* cnt, int total) {
  int i = blockIdx.x * blockDim.x + threadIdx.x;
  if (i >= total) return;
  float c = fmaxf(cnt[i >> 8], 1.f);
  hg[i] /= c;
}

// out[row] = dot(h[row,0:128], W[:,0]) + b[0]; one wave per row
__global__ __launch_bounds__(256) void k_fc3(const float* __restrict__ h,
                                             const float* __restrict__ W,
                                             const float* __restrict__ b,
                                             float* out, int rows)
{
  int wid = blockIdx.x * 8 + (threadIdx.x >> 5);
  if (wid >= rows) return;
  int lane = threadIdx.x & 31;
  float s = 0.f;
  for (int i = 0; i < 4; ++i) {
    int j = lane + i * 32;
    s += h[(size_t)wid * 128 + j] * W[j];
  }
  for (int o = 16; o; o >>= 1) s += __shfl_xor(s, o, 32);
  if (lane == 0) out[wid] = s + b[0];
}

// ---------------------------------------------------------------------------
// Host-side orchestration
// ---------------------------------------------------------------------------
static void gemm(const float* X, const float* W, const float* bias, float* out,
                 int rows, int K, int N, hipStream_t s) {
  dim3 g((N + 127) / 128, (rows + 127) / 128);
  k_gemm_bias<<<g, 256, 0, s>>>(X, W, bias, out, rows, K, N);
}

static void ln_act(const float* in, const float* sh, const float* ssum,
                   const float* res, const float* g, const float* beta,
                   float* out, int rows, int d, int act, hipStream_t s) {
  k_ln_act<<<(rows + 7) / 8, 256, 0, s>>>(in, sh, ssum, res, g, beta, out, rows, d, act);
}

struct EggcBufs {
  float *tA0, *tA1, *tA3, *tA4, *tEdge, *sumSH, *sumS;
};

static void eggc(const float* x, const float* y, const int* src, const int* dst,
                 int n, int e, const float* Wc, const float* bc,
                 const float* lng, const float* lnb,
                 const EggcBufs& B, float* x_out, float* y_out, hipStream_t s)
{
  const size_t D = 256, DD = 256 * 256;
  gemm(x, Wc + 0 * DD, bc + 0 * D, B.tA0, n, 256, 256, s);   // src gate
  gemm(x, Wc + 1 * DD, bc + 1 * D, B.tA1, n, 256, 256, s);   // dst gate
  gemm(x, Wc + 3 * DD, bc + 3 * D, B.tA3, n, 256, 256, s);   // dst update
  gemm(x, Wc + 4 * DD, bc + 4 * D, B.tA4, n, 256, 256, s);   // src update (Bh)
  gemm(y, Wc + 2 * DD, bc + 2 * D, B.tEdge, e, 256, 256, s); // edge gate
  k_gather_add3<<<(e + 7) / 8, 256, 0, s>>>(
      (const float4*)B.tA0, (const float4*)B.tA1, (const float4*)B.tEdge,
      src, dst, (float4*)B.tEdge, e);
  long long totN = (long long)n * 256;
  k_zero<<<2048, 256, 0, s>>>(B.sumSH, totN);
  k_zero<<<2048, 256, 0, s>>>(B.sumS, totN);
  k_scatter_gates<<<(e + 7) / 8, 256, 0, s>>>(
      (const float4*)B.tEdge, (const float4*)B.tA4, src, dst, B.sumSH, B.sumS, e);
  // x_out = x + silu(LN(T3 + sumSH/(sumS+eps)))
  ln_act(B.tA3, B.sumSH, B.sumS, x, lng + 0, lnb + 0, x_out, n, 256, 0, s);
  // y_out = y + silu(LN(m))
  ln_act(B.tEdge, nullptr, nullptr, y, lng + 256, lnb + 256, y_out, e, 256, 0, s);
}

extern "C" void kernel_launch(void* const* d_in, const int* in_sizes, int n_in,
                              void* d_out, int out_size, void* d_ws, size_t ws_size,
                              hipStream_t stream)
{
  // -------- inputs (pytree-alphabetical flatten; see header comment) -------
  const float* atom_feats = (const float*)d_in[0];
  const int*   dst        = (const int*)  d_in[1];
  const int*   graph_ids  = (const int*)  d_in[2];
  const int*   lg_dst     = (const int*)  d_in[3];
  const int*   lg_src     = (const int*)  d_in[4];
  const float* a1W  = (const float*)d_in[5];
  const float* a1b  = (const float*)d_in[6];
  const float* a1be = (const float*)d_in[7];
  const float* a1g  = (const float*)d_in[8];
  const float* a2W  = (const float*)d_in[9];
  const float* a2b  = (const float*)d_in[10];
  const float* a2be = (const float*)d_in[11];
  const float* a2g  = (const float*)d_in[12];
  const float* atW  = (const float*)d_in[13];
  const float* atb  = (const float*)d_in[14];
  const float* atbe = (const float*)d_in[15];
  const float* atg  = (const float*)d_in[16];
  const float* convW   = (const float*)d_in[17];  // [12,5,256,256]
  const float* convB   = (const float*)d_in[18];  // [12,5,256]
  const float* convLNg = (const float*)d_in[19];  // [12,2,256]
  const float* convLNb = (const float*)d_in[20];  // [12,2,256]
  const float* e1W  = (const float*)d_in[21];
  const float* e1b  = (const float*)d_in[22];
  const float* e1be = (const float*)d_in[23];
  const float* e1g  = (const float*)d_in[24];
  const float* e2W  = (const float*)d_in[25];
  const float* e2b  = (const float*)d_in[26];
  const float* e2be = (const float*)d_in[27];
  const float* e2g  = (const float*)d_in[28];
  const float* fcW1 = (const float*)d_in[29];
  const float* fcW2 = (const float*)d_in[30];
  const float* fcW3 = (const float*)d_in[31];
  const float* fcb1 = (const float*)d_in[32];
  const float* fcb2 = (const float*)d_in[33];
  const float* fcb3 = (const float*)d_in[34];
  const float* fcbe1 = (const float*)d_in[35];
  const float* fcbe2 = (const float*)d_in[36];
  const float* fcg1  = (const float*)d_in[37];
  const float* fcg2  = (const float*)d_in[38];
  const float* r    = (const float*)d_in[39];
  const int*   src  = (const int*)  d_in[40];
  (void)in_sizes; (void)n_in; (void)out_size; (void)ws_size;

  // -------- workspace carving (floats) -------------------------------------
  const size_t SZ_E = (size_t)N_EDGES * 256;
  const size_t SZ_T = (size_t)N_TRIP  * 256;
  const size_t SZ_N = (size_t)N_NODES * 256;

  float* p = (float*)d_ws;
  float* tA0   = p; p += SZ_E;
  float* tA1   = p; p += SZ_E;
  float* tA3   = p; p += SZ_E;
  float* tA4   = p; p += SZ_E;
  float* tEdge = p; p += SZ_T;
  float* sumSH = p; p += SZ_E;
  float* sumS  = p; p += SZ_E;
  float* xA = p; p += SZ_N;
  float* xB = p; p += SZ_N;
  float* yA = p; p += SZ_E;
  float* yB = p; p += SZ_E;
  float* zA = p; p += SZ_T;
  float* zB = p; p += SZ_T;
  float* rbfB = p; p += (size_t)N_EDGES * 80;
  float* rbfA = p; p += (size_t)N_TRIP * 40;
  float* midE = p; p += (size_t)N_EDGES * 64;
  float* midA = p; p += (size_t)N_TRIP * 64;
  float* hg   = p; p += (size_t)N_GRAPH * 256;
  float* cnt  = p; p += 64;
  float* h1   = p; p += (size_t)N_GRAPH * 256;
  float* h2   = p; p += (size_t)N_GRAPH * 128;

  EggcBufs B{tA0, tA1, tA3, tA4, tEdge, sumSH, sumS};

  // -------- input embeddings ----------------------------------------------
  gemm(atom_feats, atW, atb, xA, N_NODES, 92, 256, stream);
  ln_act(xA, nullptr, nullptr, nullptr, atg, atbe, xA, N_NODES, 256, 0, stream);

  k_bond_rbf<<<(N_EDGES + 255) / 256, 256, 0, stream>>>(r, rbfB, N_EDGES);
  gemm(rbfB, e1W, e1b, midE, N_EDGES, 80, 64, stream);
  ln_act(midE, nullptr, nullptr, nullptr, e1g, e1be, midE, N_EDGES, 64, 0, stream);
  gemm(midE, e2W, e2b, yA, N_EDGES, 64, 256, stream);
  ln_act(yA, nullptr, nullptr, nullptr, e2g, e2be, yA, N_EDGES, 256, 0, stream);

  k_angle_rbf<<<(N_TRIP + 255) / 256, 256, 0, stream>>>(r, lg_src, lg_dst, rbfA, N_TRIP);
  gemm(rbfA, a1W, a1b, midA, N_TRIP, 40, 64, stream);
  ln_act(midA, nullptr, nullptr, nullptr, a1g, a1be, midA, N_TRIP, 64, 0, stream);
  gemm(midA, a2W, a2b, zA, N_TRIP, 64, 256, stream);
  ln_act(zA, nullptr, nullptr, nullptr, a2g, a2be, zA, N_TRIP, 256, 0, stream);

  // -------- ALIGNN + GCN conv stack ----------------------------------------
  const size_t WSTRIDE = 5 * 256 * 256, BSTRIDE = 5 * 256, LSTRIDE = 2 * 256;
  float *xc = xA, *xs = xB, *yc = yA, *ys = yB, *zc = zA, *zs = zB;

  for (int i = 0; i < 4; ++i) {
    int k = 2 * i;
    eggc(xc, yc, src, dst, N_NODES, N_EDGES,
         convW + (size_t)k * WSTRIDE, convB + (size_t)k * BSTRIDE,
         convLNg + (size_t)k * LSTRIDE, convLNb + (size_t)k * LSTRIDE,
         B, xs, ys, stream);
    { float* t = xc; xc = xs; xs = t; }        // x <- x', ys holds m
    int k2 = 2 * i + 1;
    eggc(ys, zc, lg_src, lg_dst, N_EDGES, N_TRIP,
         convW + (size_t)k2 * WSTRIDE, convB + (size_t)k2 * BSTRIDE,
         convLNg + (size_t)k2 * LSTRIDE, convLNb + (size_t)k2 * LSTRIDE,
         B, yc, zs, stream);                    // new y into old y buffer
    { float* t = zc; zc = zs; zs = t; }
  }
  for (int j = 0; j < 4; ++j) {
    int k = 8 + j;
    eggc(xc, yc, src, dst, N_NODES, N_EDGES,
         convW + (size_t)k * WSTRIDE, convB + (size_t)k * BSTRIDE,
         convLNg + (size_t)k * LSTRIDE, convLNb + (size_t)k * LSTRIDE,
         B, xs, ys, stream);
    { float* t = xc; xc = xs; xs = t; }
    { float* t = yc; yc = ys; ys = t; }
  }

  // -------- readout ---------------------------------------------------------
  k_zero<<<64, 256, 0, stream>>>(hg, (long long)N_GRAPH * 256);
  k_zero<<<1, 64, 0, stream>>>(cnt, 64);
  k_pool<<<(N_NODES + 7) / 8, 256, 0, stream>>>((const float4*)xc, graph_ids, hg, cnt, N_NODES);
  k_pool_div<<<(N_GRAPH * 256 + 255) / 256, 256, 0, stream>>>(hg, cnt, N_GRAPH * 256);

  gemm(hg, fcW1, fcb1, h1, N_GRAPH, 256, 256, stream);
  ln_act(h1, nullptr, nullptr, nullptr, fcg1, fcbe1, h1, N_GRAPH, 256, 1, stream);
  gemm(h1, fcW2, fcb2, h2, N_GRAPH, 256, 128, stream);
  ln_act(h2, nullptr, nullptr, nullptr, fcg2, fcbe2, h2, N_GRAPH, 128, 1, stream);
  k_fc3<<<(N_GRAPH + 7) / 8, 256, 0, stream>>>(h2, fcW3, fcb3, (float*)d_out, N_GRAPH);
}
